// AgentCrossAttention_31774168056198
// MI455X (gfx1250) — matrix operand order repaired and checked by reading, not compiled, not run on hardware
//
#include <hip/hip_runtime.h>

// Problem constants (AgentCrossAttention): B=16, A=64, T=256, F=DK=256, H=4, DH=64
typedef __attribute__((ext_vector_type(16))) __bf16       v16bf;
typedef __attribute__((ext_vector_type(2)))  __bf16       bf16x2;
typedef __attribute__((ext_vector_type(8)))  float        v8f;
typedef __attribute__((ext_vector_type(4)))  unsigned int u32x4;
typedef __attribute__((ext_vector_type(4)))  float        f32x4;

union ABfrag { u32x4 q[2]; v16bf v; };   // 32 bytes = 16 bf16 per lane = one WMMA A/B fragment

// Native bf16 converts (RNE) -> should lower to v_cvt_pk_bf16_f32-class ops
__device__ __forceinline__ unsigned pk2(float a, float b) {
  bf16x2 v = {(__bf16)a, (__bf16)b};
  return __builtin_bit_cast(unsigned, v);
}

// ---------------------------------------------------------------------------
// Kernel 1: convert the 4 fp32 weight matrices [K=256][N=256] into bf16
// transposed layout WT[n][k] so WMMA B-fragments are contiguous per lane.
// ---------------------------------------------------------------------------
__global__ __launch_bounds__(256)
void convw_kernel(const float* __restrict__ w0, const float* __restrict__ w1,
                  const float* __restrict__ w2, const float* __restrict__ w3,
                  __bf16* __restrict__ wt) {
  const int wsel = blockIdx.x >> 8;
  const int n    = blockIdx.x & 255;
  const int k    = threadIdx.x;
  const float* w = (wsel == 0) ? w0 : (wsel == 1) ? w1 : (wsel == 2) ? w2 : w3;
  wt[wsel * 65536 + n * 256 + k] = (__bf16)w[k * 256 + n];
}

// ---------------------------------------------------------------------------
// LDS layout (bytes), 163840 total (160KB):
//  [0,      32768): xn bf16 [64 rows][256] pitch 512   (stages 0-1)
//                   then scores f32 [2 heads][64][64] + in-place attn bf16 (stage 2)
//  [32768,  65536): Q  bf16 [64][256] pitch 512
//  [65536,  98304): K  bf16 [64][256] pitch 512
//  [98304, 131072): V^T bf16 [256 dk][64 agents] pitch 128
//  [131072,163840): AO bf16 [64][256] pitch 512 (attention output, stage 2->3)
// ---------------------------------------------------------------------------
#define OFF_Q  32768
#define OFF_K  65536
#define OFF_VT 98304
#define OFF_AO 131072
#define LDS_BYTES 163840

__global__ __launch_bounds__(256)
void fused_agent_attn(const float* __restrict__ x, const int* __restrict__ fmask,
                      const __bf16* __restrict__ wT,
                      const float* __restrict__ bq, const float* __restrict__ bk,
                      const float* __restrict__ bv, const float* __restrict__ bo,
                      const float* __restrict__ gamma, const float* __restrict__ beta,
                      float* __restrict__ out) {
  __shared__ __align__(16) unsigned char smem[LDS_BYTES];
  const int tid  = threadIdx.x;
  const int lane = tid & 31;
  const int w    = tid >> 5;            // wave id 0..7
  const int b    = blockIdx.x >> 8;
  const int t    = blockIdx.x & 255;

  // ---------------- Stage 0: LayerNorm over F; xn -> LDS(bf16) + out(f32 residual)
  {
    const int l8 = lane << 3;
    const f32x4 g0 = *(const f32x4*)(gamma + l8);
    const f32x4 g1 = *(const f32x4*)(gamma + l8 + 4);
    const f32x4 e0 = *(const f32x4*)(beta + l8);
    const f32x4 e1 = *(const f32x4*)(beta + l8 + 4);
    for (int i = 0; i < 8; ++i) {
      const int a = (w << 3) + i;
      const unsigned rowoff = ((unsigned)(b * 64 + a) * 256u + (unsigned)t) << 8;
      const f32x4 v0 = *(const f32x4*)(x + rowoff + l8);
      const f32x4 v1 = *(const f32x4*)(x + rowoff + l8 + 4);
      float s  = v0.x + v0.y + v0.z + v0.w + v1.x + v1.y + v1.z + v1.w;
      float ss = v0.x*v0.x + v0.y*v0.y + v0.z*v0.z + v0.w*v0.w
               + v1.x*v1.x + v1.y*v1.y + v1.z*v1.z + v1.w*v1.w;
      #pragma unroll
      for (int o = 16; o > 0; o >>= 1) { s += __shfl_xor(s, o, 32); ss += __shfl_xor(ss, o, 32); }
      const float mu = s * (1.0f / 256.0f);
      const float rs = rsqrtf(ss * (1.0f / 256.0f) - mu * mu + 1e-5f);
      const float n0 = (v0.x - mu) * rs * g0.x + e0.x;
      const float n1 = (v0.y - mu) * rs * g0.y + e0.y;
      const float n2 = (v0.z - mu) * rs * g0.z + e0.z;
      const float n3 = (v0.w - mu) * rs * g0.w + e0.w;
      const float n4 = (v1.x - mu) * rs * g1.x + e1.x;
      const float n5 = (v1.y - mu) * rs * g1.y + e1.y;
      const float n6 = (v1.z - mu) * rs * g1.z + e1.z;
      const float n7 = (v1.w - mu) * rs * g1.w + e1.w;
      f32x4 o0 = {n0, n1, n2, n3};
      f32x4 o1 = {n4, n5, n6, n7};
      *(f32x4*)(out + rowoff + l8)     = o0;   // residual base (out = xn + proj later)
      *(f32x4*)(out + rowoff + l8 + 4) = o1;
      u32x4 pu = {pk2(n0, n1), pk2(n2, n3), pk2(n4, n5), pk2(n6, n7)};
      *(u32x4*)(smem + (a << 9) + (lane << 4)) = pu;
    }
  }
  __threadfence_block();
  __syncthreads();

  // Per-wave WMMA geometry (wave32; A and B share a contiguous K order):
  const int m0    = (w & 3) << 4;           // M tile (agent rows)
  const int nhalf = w >> 2;                 // which half of the 16 N-tiles
  const int arow  = m0 + (lane & 15);       // A: both lane halves cover rows 0..15
  const int ac0   = (lane & 16) ? 16 : 0;   // A: 16 contiguous K per lane
  const int bc0   = (lane & 16) ? 16 : 0;   // B: 16 contiguous K per lane
  const int dbase = (lane & 16) ? 8 : 0;    // C/D: row = m0 + dbase + vgpr

  // ---------------- Stage 1: Q/K/V = xn @ W + b  (bf16 WMMA, f32 acc)
  {
    ABfrag af[8];                           // A fragments reused for all 3 weights
    #pragma unroll
    for (int kt = 0; kt < 8; ++kt) {
      const unsigned char* p = smem + (arow << 9) + ((kt * 32 + ac0) << 1);
      af[kt].q[0] = *(const u32x4*)p;
      af[kt].q[1] = *(const u32x4*)(p + 16);
    }
    __builtin_prefetch((const void*)wT, 0, 1);
    #pragma unroll
    for (int wsel = 0; wsel < 3; ++wsel) {
      const __bf16* wp   = wT + wsel * 65536;
      const float*  bias = (wsel == 0) ? bq : (wsel == 1) ? bk : bv;
      for (int ntl = 0; ntl < 8; ++ntl) {
        const int ncol = ((nhalf * 8 + ntl) << 4) + (lane & 15);
        v8f acc = {0.f, 0.f, 0.f, 0.f, 0.f, 0.f, 0.f, 0.f};
        #pragma unroll
        for (int kt = 0; kt < 8; ++kt) {
          ABfrag bf_;
          const __bf16* p = wp + ncol * 256 + kt * 32 + bc0;
          bf_.q[0] = *(const u32x4*)p;
          bf_.q[1] = *(const u32x4*)(p + 8);
          acc = __builtin_amdgcn_wmma_f32_16x16x32_bf16(false, af[kt].v, false, bf_.v,
                                                        (short)0, acc, false, false);
        }
        const float bb = bias[ncol];
        if (wsel < 2) {                      // Q,K stored row-major [agent][256]
          unsigned char* qk = smem + ((wsel == 0) ? OFF_Q : OFF_K);
          #pragma unroll
          for (int r = 0; r < 8; ++r)
            *(__bf16*)(qk + ((m0 + dbase + r) << 9) + (ncol << 1)) = (__bf16)(acc[r] + bb);
        } else {                             // V stored transposed [dk][agent], packed store
          u32x4 pv = {pk2(acc[0]+bb, acc[1]+bb), pk2(acc[2]+bb, acc[3]+bb),
                      pk2(acc[4]+bb, acc[5]+bb), pk2(acc[6]+bb, acc[7]+bb)};
          *(u32x4*)(smem + OFF_VT + (ncol << 7) + ((m0 + dbase) << 1)) = pv;
        }
      }
    }
  }
  __syncthreads();

  // ---------------- Stage 2: attention, two heads per pass (scores reuse xn region)
  for (int hp = 0; hp < 2; ++hp) {
    for (int j = 0; j < 4; ++j) {           // 32 score tiles / 8 waves
      const int tt = (w << 2) + j;
      const int hl = tt >> 4;
      const int h  = hp * 2 + hl;
      const int sm = (tt >> 2) & 3;
      const int sn = tt & 3;
      const int qrow = (sm << 4) + (lane & 15);
      const int krow = (sn << 4) + (lane & 15);
      v8f acc = {0.f, 0.f, 0.f, 0.f, 0.f, 0.f, 0.f, 0.f};
      #pragma unroll
      for (int kt = 0; kt < 2; ++kt) {      // DH=64 -> 2 K-tiles
        ABfrag afr, bfr;
        const unsigned char* qa = smem + OFF_Q + (qrow << 9) + ((h * 64 + kt * 32 + ac0) << 1);
        afr.q[0] = *(const u32x4*)qa;  afr.q[1] = *(const u32x4*)(qa + 16);
        const unsigned char* kb = smem + OFF_K + (krow << 9) + ((h * 64 + kt * 32 + bc0) << 1);
        bfr.q[0] = *(const u32x4*)kb;  bfr.q[1] = *(const u32x4*)(kb + 16);
        acc = __builtin_amdgcn_wmma_f32_16x16x32_bf16(false, afr.v, false, bfr.v,
                                                      (short)0, acc, false, false);
      }
      unsigned char* sc = smem + (hl << 14);                 // [query][key] f32
      #pragma unroll
      for (int r = 0; r < 8; ++r)
        *(float*)(sc + ((sm * 16 + dbase + r) << 8) + ((sn * 16 + (lane & 15)) << 2))
            = acc[r] * 0.125f;                               // 1/sqrt(DH)
    }
    __syncthreads();

    if (tid < 128) {                        // masked softmax: one row per thread
      const int hl = tid >> 6;
      const int qi = tid & 63;
      float* row = (float*)(smem + (hl << 14) + (qi << 8));
      float mx = -__builtin_inff();
      for (int k = 0; k < 64; ++k) {
        const int   mv = fmask[((b * 64 + k) << 8) + t];
        const float sv = mv ? row[k] : -__builtin_inff();
        row[k] = sv;
        mx = fmaxf(mx, sv);
      }
      __bf16* arow16 = (__bf16*)row;                         // attn bf16 in-place
      if (mx > -__builtin_inff()) {
        float sum = 0.f;
        for (int k = 0; k < 64; ++k) { const float e = __expf(row[k] - mx); row[k] = e; sum += e; }
        const float rinv = 1.0f / sum;
        for (int k = 0; k < 64; ++k) arow16[k] = (__bf16)(row[k] * rinv);
      } else {
        for (int k = 0; k < 64; ++k) arow16[k] = (__bf16)0.f; // no valid agent -> zeros
      }
    }
    __syncthreads();

    for (int j = 0; j < 4; ++j) {           // out_h = attn @ V  -> AO
      const int tt = (w << 2) + j;
      const int hl = tt >> 4;
      const int h  = hp * 2 + hl;
      const int sm = (tt >> 2) & 3;
      const int sn = tt & 3;
      const int prow = (sm << 4) + (lane & 15);
      const int vdk  = h * 64 + (sn << 4) + (lane & 15);
      v8f acc = {0.f, 0.f, 0.f, 0.f, 0.f, 0.f, 0.f, 0.f};
      #pragma unroll
      for (int kt = 0; kt < 2; ++kt) {
        ABfrag afr, bfr;
        const unsigned char* pa = smem + (hl << 14) + (prow << 8) + ((kt * 32 + ac0) << 1);
        afr.q[0] = *(const u32x4*)pa;  afr.q[1] = *(const u32x4*)(pa + 16);
        const unsigned char* vb = smem + OFF_VT + (vdk << 7) + ((kt * 32 + bc0) << 1);
        bfr.q[0] = *(const u32x4*)vb;  bfr.q[1] = *(const u32x4*)(vb + 16);
        acc = __builtin_amdgcn_wmma_f32_16x16x32_bf16(false, afr.v, false, bfr.v,
                                                      (short)0, acc, false, false);
      }
      #pragma unroll
      for (int r = 0; r < 8; ++r)
        *(__bf16*)(smem + OFF_AO + ((sm * 16 + dbase + r) << 9) + (vdk << 1))
            = (__bf16)acc[r];
    }
    __syncthreads();
  }

  // ---------------- Stage 3: out += AO @ Wo + bo  (residual xn already in out)
  {
    ABfrag af[8];
    #pragma unroll
    for (int kt = 0; kt < 8; ++kt) {
      const unsigned char* p = smem + OFF_AO + (arow << 9) + ((kt * 32 + ac0) << 1);
      af[kt].q[0] = *(const u32x4*)p;
      af[kt].q[1] = *(const u32x4*)(p + 16);
    }
    const __bf16* wo = wT + 3 * 65536;
    __builtin_prefetch((const void*)wo, 0, 1);
    for (int ntl = 0; ntl < 8; ++ntl) {
      const int ncol = ((nhalf * 8 + ntl) << 4) + (lane & 15);
      v8f acc = {0.f, 0.f, 0.f, 0.f, 0.f, 0.f, 0.f, 0.f};
      #pragma unroll
      for (int kt = 0; kt < 8; ++kt) {
        ABfrag bf_;
        const __bf16* p = wo + ncol * 256 + kt * 32 + bc0;
        bf_.q[0] = *(const u32x4*)p;
        bf_.q[1] = *(const u32x4*)(p + 8);
        acc = __builtin_amdgcn_wmma_f32_16x16x32_bf16(false, af[kt].v, false, bf_.v,
                                                      (short)0, acc, false, false);
      }
      const float bb = bo[ncol];
      #pragma unroll
      for (int r = 0; r < 8; ++r) {
        const int a = m0 + dbase + r;
        float* op = out + (((unsigned)(b * 64 + a) * 256u + (unsigned)t) << 8) + ncol;
        *op = *op + acc[r] + bb;
      }
    }
  }
}

// ---------------------------------------------------------------------------
extern "C" void kernel_launch(void* const* d_in, const int* in_sizes, int n_in,
                              void* d_out, int out_size, void* d_ws, size_t ws_size,
                              hipStream_t stream) {
  const float* x     = (const float*)d_in[0];
  const int*   fmask = (const int*)d_in[1];
  const float* Wq    = (const float*)d_in[2];
  const float* bq    = (const float*)d_in[3];
  const float* Wk    = (const float*)d_in[4];
  const float* bk    = (const float*)d_in[5];
  const float* Wv    = (const float*)d_in[6];
  const float* bv    = (const float*)d_in[7];
  const float* Wo    = (const float*)d_in[8];
  const float* bo    = (const float*)d_in[9];
  const float* gamma = (const float*)d_in[10];
  const float* beta  = (const float*)d_in[11];
  float* out = (float*)d_out;
  __bf16* wt = (__bf16*)d_ws;   // 4 * 256*256 bf16 = 512KB

  convw_kernel<<<1024, 256, 0, stream>>>(Wq, Wk, Wv, Wo, wt);
  fused_agent_attn<<<16 * 256, 256, 0, stream>>>(x, fmask, (const __bf16*)d_ws,
                                                 bq, bk, bv, bo, gamma, beta, out);
}
